// ChannelAttentionModule_13245679141616
// MI455X (gfx1250) — matrix operand order repaired and checked by reading, not compiled, not run on hardware
//
#include <hip/hip_runtime.h>
#include <hip/hip_bf16.h>

// CDNA5 / gfx1250, wave32. BF16 WMMA (16x16x32, f32 accum) for both GEMMs.
typedef __attribute__((ext_vector_type(16))) __bf16 v16bf;
typedef __attribute__((ext_vector_type(8)))  float  v8f;

#define B_  2
#define C_  128
#define N_  (16*96*96)      // 147456
#define SLICE 1152          // N per split-K workgroup (128 slices per batch)
#define KT 64               // n-elements staged into LDS per iteration
#define LSTR 72             // padded LDS row stride, k_energy (bf16 elems)
#define QSTR 136            // padded LDS row stride, k_out (bf16 elems)

// ---------- K0: x(f32, [B,C,N]) -> q_t(bf16, [B,N,C]) via LDS transpose ----
__global__ __launch_bounds__(256) void k_convert(const float* __restrict__ x,
                                                 __bf16* __restrict__ qt) {
  __shared__ __align__(16) __bf16 tile[64][40];   // 64 n x 32 c (+pad)
  const int t  = threadIdx.x;
  const int n0 = blockIdx.x * 64;
  const int c0 = blockIdx.y * 32;
  const int b  = blockIdx.z;
  #pragma unroll
  for (int rr = 0; rr < 2; ++rr) {
    int cl = rr * 16 + (t >> 4);          // 0..31 channel-in-tile
    int nl = (t & 15) * 4;                // 0..60 n-in-tile
    size_t base = ((size_t)(b * C_ + c0 + cl)) * N_ + n0 + nl;
    float4 f = *(const float4*)(x + base);
    tile[nl + 0][cl] = (__bf16)f.x; tile[nl + 1][cl] = (__bf16)f.y;
    tile[nl + 2][cl] = (__bf16)f.z; tile[nl + 3][cl] = (__bf16)f.w;
  }
  __syncthreads();
  int nl   = t >> 2;                      // 0..63
  int part = (t & 3) * 8;                 // 0,8,16,24
  uint4 v = *(const uint4*)&tile[nl][part];         // 16B aligned (80B rows)
  size_t base = ((size_t)b * N_ + n0 + nl) * C_ + c0 + part;
  *(uint4*)(qt + base) = v;
}

// ---------- K1: energy[b] += q_slice * q_slice^T (split-K, bf16 WMMA) ------
__global__ __launch_bounds__(256) void k_energy(const float* __restrict__ x,
                                                float* __restrict__ energy) {
  __shared__ __align__(16) __bf16 lds[C_ * LSTR];   // 128 x 64 (+pad) bf16
  const int t    = threadIdx.x;
  const int b    = blockIdx.y;
  const int n0   = blockIdx.x * SLICE;
  const int lane = t & 31;
  const int half = lane >> 4;
  const int l16  = lane & 15;
  const int m0   = (t >> 5) * 16;         // this wave's c1 block

  v8f acc[8] = {};                        // 8 c2-tiles of 16x16 f32
  const float* gx = x + (size_t)b * C_ * N_;
  const int srow = t >> 1, scb = (t & 1) * 32;       // staging assignment

  for (int it = 0; it < SLICE / KT; ++it) {
    const int nbase = n0 + it * KT;
    { // stage x -> bf16 LDS: 128 rows x 64 cols
      const float* src = gx + (size_t)srow * N_ + nbase + scb;
      __bf16* dst = &lds[srow * LSTR + scb];
      #pragma unroll
      for (int j = 0; j < 8; ++j) {
        float4 f = *(const float4*)(src + j * 4);
        union { __bf16 h[4]; uint2 u; } p;
        p.h[0] = (__bf16)f.x; p.h[1] = (__bf16)f.y;
        p.h[2] = (__bf16)f.z; p.h[3] = (__bf16)f.w;
        *(uint2*)(dst + j * 4) = p.u;
      }
      if (it + 1 < SLICE / KT)            // warm L2/L0 for next K-slice
        __builtin_prefetch(src + KT, 0, 3);
    }
    __syncthreads();
    #pragma unroll
    for (int kk = 0; kk < KT; kk += 32) {
      union F16 { v16bf v; uint4 h[2]; };
      // A fragment: A[m][k] = q[m0+m][nbase+kk+k]  (16-bit A 16x32 layout)
      F16 a;
      const __bf16* ar = &lds[(m0 + l16) * LSTR + kk + 8 * half];
      a.h[0] = *(const uint4*)(ar);            // K = kk+8h .. +7
      a.h[1] = *(const uint4*)(ar + 16);       // K = kk+16+8h .. +7
      // Preload ALL 8 B fragments into distinct regs, then chain WMMAs:
      // waits become descending dscnt thresholds instead of 8x dscnt==0.
      F16 bfr[8];
      #pragma unroll
      for (int j2 = 0; j2 < 8; ++j2) {
        const __bf16* br = &lds[(j2 * 16 + l16) * LSTR + kk + 8 * half];
        bfr[j2].h[0] = *(const uint4*)(br);
        bfr[j2].h[1] = *(const uint4*)(br + 16);
      }
      #pragma unroll
      for (int j2 = 0; j2 < 8; ++j2)
        acc[j2] = __builtin_amdgcn_wmma_f32_16x16x32_bf16(
            false, a.v, false, bfr[j2].v, (short)0, acc[j2], false, false);
    }
    __syncthreads();
  }
  // D layout: lane -> N=l16, VGPR r -> M=r+8*half
  float* e = energy + (size_t)b * C_ * C_;
  #pragma unroll
  for (int j2 = 0; j2 < 8; ++j2)
    #pragma unroll
    for (int r = 0; r < 8; ++r)
      unsafeAtomicAdd(&e[(m0 + r + 8 * half) * C_ + j2 * 16 + l16], acc[j2][r]);
}

// ---------- K2: softmax(rowmax - e) == exp(rowmin - e)/sum; one wave/row ----
__global__ __launch_bounds__(256) void k_softmax(const float* __restrict__ energy,
                                                 __bf16* __restrict__ attn) {
  const int row  = blockIdx.x * 8 + (threadIdx.x >> 5);
  const int lane = threadIdx.x & 31;
  const float* e = energy + (size_t)row * C_;
  float4 v = *(const float4*)(e + lane * 4);
  float mn = fminf(fminf(v.x, v.y), fminf(v.z, v.w));
  #pragma unroll
  for (int s = 16; s; s >>= 1) mn = fminf(mn, __shfl_xor(mn, s, 32));
  float e0 = __expf(mn - v.x), e1 = __expf(mn - v.y);
  float e2 = __expf(mn - v.z), e3 = __expf(mn - v.w);
  float sum = e0 + e1 + e2 + e3;
  #pragma unroll
  for (int s = 16; s; s >>= 1) sum += __shfl_xor(sum, s, 32);
  float inv = 1.0f / sum;
  union { __bf16 h[4]; uint2 u; } p;
  p.h[0] = (__bf16)(e0 * inv); p.h[1] = (__bf16)(e1 * inv);
  p.h[2] = (__bf16)(e2 * inv); p.h[3] = (__bf16)(e3 * inv);
  *(uint2*)(attn + (size_t)row * C_ + lane * 4) = p.u;
}

// ---------- K3: out = x + gamma * (attn @ q), bf16 WMMA, K=128 -------------
// qt supertile (128 n x 128 c, 32KB) staged once per WG via async-to-LDS:
// shared by all 8 waves (B fragments are wave-invariant).
__global__ __launch_bounds__(256) void k_out(const float* __restrict__ x,
                                             const __bf16* __restrict__ qt,
                                             const __bf16* __restrict__ attn,
                                             const float* __restrict__ gamma,
                                             float* __restrict__ out) {
  __shared__ __align__(16) __bf16 bqt[128 * QSTR];  // 128 rows, padded
  const int b    = blockIdx.y;
  const int nsup = blockIdx.x * 128;
  const int t    = threadIdx.x;
  const int lane = t & 31;
  const int half = lane >> 4;
  const int l16  = lane & 15;
  const int m0   = (t >> 5) * 16;         // c1 block
  const float g  = gamma[0];

  { // async copy: qt[nsup+row][0..127] -> bqt[row][0..127]; 128B per thread
    const int row = t >> 1, cb = (t & 1) * 64;
    const __bf16* src = qt + ((size_t)b * N_ + nsup + row) * C_ + cb;
    unsigned ldsa = (unsigned)(size_t)&bqt[row * QSTR + cb];
    asm volatile("global_load_async_to_lds_b128 %0, %1, off offset:0"
                 :: "v"(ldsa), "v"(src) : "memory");
    asm volatile("global_load_async_to_lds_b128 %0, %1, off offset:16"
                 :: "v"(ldsa), "v"(src) : "memory");
    asm volatile("global_load_async_to_lds_b128 %0, %1, off offset:32"
                 :: "v"(ldsa), "v"(src) : "memory");
    asm volatile("global_load_async_to_lds_b128 %0, %1, off offset:48"
                 :: "v"(ldsa), "v"(src) : "memory");
    asm volatile("global_load_async_to_lds_b128 %0, %1, off offset:64"
                 :: "v"(ldsa), "v"(src) : "memory");
    asm volatile("global_load_async_to_lds_b128 %0, %1, off offset:80"
                 :: "v"(ldsa), "v"(src) : "memory");
    asm volatile("global_load_async_to_lds_b128 %0, %1, off offset:96"
                 :: "v"(ldsa), "v"(src) : "memory");
    asm volatile("global_load_async_to_lds_b128 %0, %1, off offset:112"
                 :: "v"(ldsa), "v"(src) : "memory");
    asm volatile("s_wait_asynccnt 0" ::: "memory");
  }

  union F { v16bf v; uint4 h[2]; };
  F a[4];                                 // attn[m0..m0+15][0..127] as 4 frags
  const __bf16* arow = attn + ((size_t)b * C_ + m0 + l16) * C_;
  #pragma unroll
  for (int k = 0; k < 4; ++k) {
    a[k].h[0] = *(const uint4*)(arow + k * 32 + 8 * half);
    a[k].h[1] = *(const uint4*)(arow + k * 32 + 16 + 8 * half);
  }
  __syncthreads();                        // LDS tile visible to all waves

  #pragma unroll
  for (int j = 0; j < 8; ++j) {
    const int nt = nsup + j * 16;
    // B[k][n] = q[c2=k][nt+n] = bqt[j*16+n][k]: two 16B chunks at k=8h, 16+8h
    const __bf16* brow = &bqt[(j * 16 + l16) * QSTR + 8 * half];
    F bf[4];
    #pragma unroll
    for (int k = 0; k < 4; ++k) {          // all loads first, then WMMA chain
      bf[k].h[0] = *(const uint4*)(brow + k * 32);
      bf[k].h[1] = *(const uint4*)(brow + k * 32 + 16);
    }
    v8f acc = {};
    #pragma unroll
    for (int k = 0; k < 4; ++k)
      acc = __builtin_amdgcn_wmma_f32_16x16x32_bf16(
          false, a[k].v, false, bf[k].v, (short)0, acc, false, false);
    #pragma unroll
    for (int r = 0; r < 8; ++r) {
      size_t idx = ((size_t)b * C_ + m0 + r + 8 * half) * N_ + nt + l16;
      out[idx] = x[idx] + g * acc[r];
    }
  }
}

extern "C" void kernel_launch(void* const* d_in, const int* in_sizes, int n_in,
                              void* d_out, int out_size, void* d_ws, size_t ws_size,
                              hipStream_t stream) {
  const float* x     = (const float*)d_in[0];
  const float* gamma = (const float*)d_in[1];
  float* out = (float*)d_out;

  char* ws = (char*)d_ws;
  const size_t qtBytes = (size_t)B_ * N_ * C_ * sizeof(__bf16);       // ~75.5 MB
  const size_t enBytes = (size_t)B_ * C_ * C_ * sizeof(float);        // 128 KB
  __bf16* qt     = (__bf16*)ws;
  float*  energy = (float*)(ws + qtBytes);
  __bf16* attn   = (__bf16*)(ws + qtBytes + enBytes);

  hipMemsetAsync(energy, 0, enBytes, stream);
  k_convert<<<dim3(N_ / 64, C_ / 32, B_), 256, 0, stream>>>(x, qt);
  k_energy <<<dim3(N_ / SLICE, B_),      256, 0, stream>>>(x, energy);
  k_softmax<<<dim3(B_ * C_ / 8),         256, 0, stream>>>(energy, attn);
  k_out    <<<dim3(N_ / 128, B_),        256, 0, stream>>>(x, qt, attn, gamma, out);
}
